// mysat_net_68504728371325
// MI455X (gfx1250) — matrix operand order repaired and checked by reading, not compiled, not run on hardware
//
#include <hip/hip_runtime.h>
#include <hip/hip_bf16.h>
#include <math.h>

// ---------------------------------------------------------------------------
// NeuroSAT-style GNN on MI455X (gfx1250).
// All matmuls run through a WMMA (v_wmma_f32_16x16x32_f16) tiled GEMM with
// fp32 accumulate, double-buffered LDS staging so global loads of the next
// K-slab overlap the WMMAs of the current one. Mask Ms is converted once per
// call to f16 (128MB -> fits the 192MB L2), so the 32 big einsum passes
// stream their A operand from L2.
// ---------------------------------------------------------------------------

typedef __attribute__((ext_vector_type(16))) _Float16 v16h;
typedef __attribute__((ext_vector_type(8)))  _Float16 v8h;
typedef __attribute__((ext_vector_type(4)))  _Float16 v4h;
typedef __attribute__((ext_vector_type(8)))  float    v8f;
typedef __attribute__((ext_vector_type(4)))  float    vf4;

#define HDIM   128
#define NBATCH 8
#define LDIM   2048
#define CDIM   4096
#define TSTEPS 16

template <typename T> struct rv4;
template <> struct rv4<float>    { typedef vf4 t; };
template <> struct rv4<_Float16> { typedef v4h t; };

__device__ __forceinline__ float sigmoidf_(float x) { return 1.0f / (1.0f + expf(-x)); }

__device__ __forceinline__ v4h cvt4(vf4 x) {
    v4h r;
    r.x = (_Float16)x.x; r.y = (_Float16)x.y;
    r.z = (_Float16)x.z; r.w = (_Float16)x.w;
    return r;
}
__device__ __forceinline__ v4h cvt4(v4h x) { return x; }

__device__ __forceinline__ v4h load4h(const float* p)    { return cvt4(*(const vf4*)p); }
__device__ __forceinline__ v4h load4h(const _Float16* p) { return *(const v4h*)p; }

// ---------------------------------------------------------------------------
// WMMA GEMM:  C[m,n] = sum_k A(m,k)*B(k,n) + bias[n]   (optional ReLU)
//   A(m,k) = TRANSA ? A[k*lda+m] : A[m*lda+k]   (TA = float or _Float16)
//   B(k,n) = TRANSB ? B[n*ldb+k] : B[k*ldb+n]   (float)
// Block = 128 threads (4 waves), tile 64x64, K-step 32; each wave owns a
// 32x32 quadrant = 2x2 v_wmma_f32_16x16x32_f16 per K-step. LDS tiles are
// double-buffered; raw next-slab values are parked in VGPRs during compute.
// Requires M%64==0, N%64==0, K%32==0, lda/ldb%4==0 (true for every GEMM here).
// ---------------------------------------------------------------------------
template <typename TA, bool TRANSA, bool TRANSB, bool RELU>
__global__ __launch_bounds__(128)
void wmma_gemm(const TA* __restrict__ Ag, const float* __restrict__ Bg,
               const float* __restrict__ bias, float* __restrict__ Cg,
               int M, int N, int K, int lda, int ldb, int ldc,
               long sA, long sB, long sC)
{
    const TA*    A = Ag + (long)blockIdx.z * sA;
    const float* B = Bg + (long)blockIdx.z * sB;
    float*       C = Cg + (long)blockIdx.z * sC;

    const int rowBase = blockIdx.y * 64;
    const int colBase = blockIdx.x * 64;
    const int tid  = threadIdx.x;
    const int lane = tid & 31;
    const int wave = tid >> 5;
    const int wm   = (wave >> 1) * 32;   // wave row offset in tile
    const int wn   = (wave & 1) * 32;    // wave col offset in tile
    const int l16  = lane & 15;          // row (A) / col (B,C) within 16x16
    const int hsel = lane >> 4;          // half-select per ISA 7.12.2

    __shared__ __align__(32) _Float16 Ash[2][64 * 32];   // [row][k]
    __shared__ __align__(32) _Float16 Bsh[2][64 * 32];   // [col][k] (n-major)

    typedef typename rv4<TA>::t rawA_t;
    rawA_t ra[4];
    vf4    rb[4];

    // ---- register-staging helpers (loads only; no LDS dependency) ----
    auto loadA = [&](int k0) {
#pragma unroll
        for (int s = 0; s < 4; ++s) {
            int v = tid + 128 * s;
            if (!TRANSA) {
                int r = v >> 3, kk = (v & 7) * 4;
                ra[s] = *(const rawA_t*)&A[(long)(rowBase + r) * lda + (k0 + kk)];
            } else {
                int kk = v >> 4, rg = v & 15;
                ra[s] = *(const rawA_t*)&A[(long)(k0 + kk) * lda + (rowBase + rg * 4)];
            }
        }
    };
    auto loadB = [&](int k0) {
#pragma unroll
        for (int s = 0; s < 4; ++s) {
            int v = tid + 128 * s;
            if (TRANSB) {
                int cn = v >> 3, kk = (v & 7) * 4;
                rb[s] = *(const vf4*)&B[(long)(colBase + cn) * ldb + (k0 + kk)];
            } else {
                int kk = v >> 4, ng = v & 15;
                rb[s] = *(const vf4*)&B[(long)(k0 + kk) * ldb + (colBase + ng * 4)];
            }
        }
    };
    // ---- convert + store helpers ----
    auto storeA = [&](int p) {
#pragma unroll
        for (int s = 0; s < 4; ++s) {
            int v = tid + 128 * s;
            v4h x = cvt4(ra[s]);
            if (!TRANSA) {
                int r = v >> 3, kk = (v & 7) * 4;
                *(v4h*)&Ash[p][r * 32 + kk] = x;
            } else {
                int kk = v >> 4, rg = v & 15;
                Ash[p][(rg * 4 + 0) * 32 + kk] = x.x;
                Ash[p][(rg * 4 + 1) * 32 + kk] = x.y;
                Ash[p][(rg * 4 + 2) * 32 + kk] = x.z;
                Ash[p][(rg * 4 + 3) * 32 + kk] = x.w;
            }
        }
    };
    auto storeB = [&](int p) {
#pragma unroll
        for (int s = 0; s < 4; ++s) {
            int v = tid + 128 * s;
            v4h x = cvt4(rb[s]);
            if (TRANSB) {
                int cn = v >> 3, kk = (v & 7) * 4;
                *(v4h*)&Bsh[p][cn * 32 + kk] = x;
            } else {
                int kk = v >> 4, ng = v & 15;
                Bsh[p][(ng * 4 + 0) * 32 + kk] = x.x;
                Bsh[p][(ng * 4 + 1) * 32 + kk] = x.y;
                Bsh[p][(ng * 4 + 2) * 32 + kk] = x.z;
                Bsh[p][(ng * 4 + 3) * 32 + kk] = x.w;
            }
        }
    };

    v8f acc[2][2] = {};

    // Prologue: fill buffer 0.
    loadA(0); loadB(0);
    storeA(0); storeB(0);

    int p = 0;
    for (int k0 = 0; k0 < K; k0 += 32) {
        __syncthreads();
        const bool hasNext = (k0 + 32) < K;
        if (hasNext) { loadA(k0 + 32); loadB(k0 + 32); }   // in flight during WMMA
        if (k0 + 64 < K) {                                  // global_prefetch_b8
            const int r = tid & 63;
            __builtin_prefetch(TRANSA ? (const void*)&A[(long)(k0 + 64 + (tid >> 6) * 16) * lda + rowBase + r]
                                      : (const void*)&A[(long)(rowBase + r) * lda + (k0 + 64)], 0, 0);
            __builtin_prefetch(TRANSB ? (const void*)&B[(long)(colBase + r) * ldb + (k0 + 64)]
                                      : (const void*)&B[(long)(k0 + 64 + (tid >> 6) * 16) * ldb + colBase + r], 0, 0);
        }

        // A fragment (16x32): lanes0-15 K=[0..7]+[16..23]; lanes16-31 K=[8..15]+[24..31]
        v16h afrag[2], bfrag[2];
#pragma unroll
        for (int i = 0; i < 2; ++i) {
            int r = wm + 16 * i + l16;
            v8h lo = *(const v8h*)&Ash[p][r * 32 + 8 * hsel];
            v8h hi = *(const v8h*)&Ash[p][r * 32 + 16 + 8 * hsel];
            afrag[i] = __builtin_shufflevector(lo, hi,
                         0,1,2,3,4,5,6,7,8,9,10,11,12,13,14,15);
        }
        // B fragment (32x16): lane -> col, K order same as A: [0..7]+[16..23] / [8..15]+[24..31]
#pragma unroll
        for (int j = 0; j < 2; ++j) {
            int n = wn + 16 * j + l16;
            v8h lo = *(const v8h*)&Bsh[p][n * 32 + 8 * hsel];
            v8h hi = *(const v8h*)&Bsh[p][n * 32 + 16 + 8 * hsel];
            bfrag[j] = __builtin_shufflevector(lo, hi,
                         0,1,2,3,4,5,6,7,8,9,10,11,12,13,14,15);
        }
#pragma unroll
        for (int i = 0; i < 2; ++i)
#pragma unroll
            for (int j = 0; j < 2; ++j)
                acc[i][j] = __builtin_amdgcn_wmma_f32_16x16x32_f16(
                    false, afrag[i], false, bfrag[j],
                    (short)0, acc[i][j], false, false);

        if (hasNext) { storeA(p ^ 1); storeB(p ^ 1); }
        p ^= 1;
    }

    // Epilogue. C layout: VGPR r -> M = r + 8*hsel, lane&15 -> N.
#pragma unroll
    for (int i = 0; i < 2; ++i) {
#pragma unroll
        for (int j = 0; j < 2; ++j) {
            int col = colBase + wn + 16 * j + l16;
            float bv = bias ? bias[col] : 0.0f;
#pragma unroll
            for (int r = 0; r < 8; ++r) {
                int row = rowBase + wm + 16 * i + 8 * hsel + r;
                float v = acc[i][j][r] + bv;
                if (RELU) v = fmaxf(v, 0.0f);
                C[(long)row * ldc + col] = v;
            }
        }
    }
}

// GRU gate fusion: gi,gh are (rows,384) WITH biases folded in; h is (rows,128).
__global__ void gru_combine(const float* __restrict__ gi,
                            const float* __restrict__ gh,
                            float* __restrict__ h, int rows)
{
    long idx = (long)blockIdx.x * blockDim.x + threadIdx.x;
    long total = (long)rows * HDIM;
    if (idx >= total) return;
    long row = idx >> 7;
    int  col = (int)(idx & 127);
    const float* gir = gi + row * 384;
    const float* ghr = gh + row * 384;
    float r = sigmoidf_(gir[col]       + ghr[col]);
    float z = sigmoidf_(gir[128 + col] + ghr[128 + col]);
    float n = tanhf(gir[256 + col] + r * ghr[256 + col]);
    float hv = h[idx];
    h[idx] = (1.0f - z) * n + z * hv;
}

__global__ void fill_bcast(float* __restrict__ out, const float* __restrict__ src, long total)
{
    long i = (long)blockIdx.x * blockDim.x + threadIdx.x;
    if (i < total) out[i] = src[i & (HDIM - 1)];
}

__global__ void f32_to_f16(const float* __restrict__ in, _Float16* __restrict__ out, long n)
{
    long i4 = ((long)blockIdx.x * blockDim.x + threadIdx.x) * 4;
    if (i4 < n) *(v4h*)&out[i4] = load4h(&in[i4]);
}

// gi_base[n] = Lu_bih[n] + sum_k Linit[k] * Lu_Wih[n, k]   (FL == bcast(Linit))
__global__ void compute_gibase(const float* __restrict__ Linit,
                               const float* __restrict__ Wih,
                               const float* __restrict__ bih,
                               float* __restrict__ out)
{
    int n = blockIdx.x * blockDim.x + threadIdx.x;
    if (n >= 384) return;
    float s = bih[n];
    for (int k = 0; k < HDIM; ++k) s += Linit[k] * Wih[n * 256 + k];
    out[n] = s;
}

// lvotes[m] = b2 + sum_k h1[m,k] * W2[k]   (N==1 projection)
__global__ void lvotes_kernel(const float* __restrict__ h1,
                              const float* __restrict__ W2,
                              const float* __restrict__ b2,
                              float* __restrict__ out, int rows)
{
    int m = blockIdx.x * blockDim.x + threadIdx.x;
    if (m >= rows) return;
    float s = b2[0];
    const float* r = h1 + (long)m * 256;
    for (int k = 0; k < 256; ++k) s += r[k] * W2[k];
    out[m] = s;
}

__global__ void readout_mean(const float* __restrict__ lv, float* __restrict__ y)
{
    __shared__ float red[256];
    int b = blockIdx.x;
    float s = 0.0f;
    for (int i = threadIdx.x; i < LDIM; i += 256) s += lv[b * LDIM + i];
    red[threadIdx.x] = s;
    __syncthreads();
    for (int o = 128; o > 0; o >>= 1) {
        if (threadIdx.x < o) red[threadIdx.x] += red[threadIdx.x + o];
        __syncthreads();
    }
    if (threadIdx.x == 0) y[b] = sigmoidf_(red[0] / (float)LDIM);
}

template <typename TA, bool TRANSA, bool TRANSB, bool RELU>
static inline void launch_gemm(hipStream_t s, const TA* A, const float* B,
                               const float* bias, float* C,
                               int M, int N, int K, int lda, int ldb, int ldc,
                               long sA = 0, long sB = 0, long sC = 0, int batch = 1)
{
    dim3 grid(N / 64, M / 64, batch);
    wmma_gemm<TA, TRANSA, TRANSB, RELU><<<grid, 128, 0, s>>>(
        A, B, bias, C, M, N, K, lda, ldb, ldc, sA, sB, sC);
}

extern "C" void kernel_launch(void* const* d_in, const int* in_sizes, int n_in,
                              void* d_out, int out_size, void* d_ws, size_t ws_size,
                              hipStream_t stream)
{
    (void)in_sizes; (void)n_in; (void)out_size;
    const float* Ms    = (const float*)d_in[0];
    const float* Linit = (const float*)d_in[1];
    const float* Cinit = (const float*)d_in[2];
    const float* LmsgW = (const float*)d_in[3];
    const float* Lmsgb = (const float*)d_in[4];
    const float* CmsgW = (const float*)d_in[5];
    const float* Cmsgb = (const float*)d_in[6];
    const float* CuWih = (const float*)d_in[7];
    const float* CuWhh = (const float*)d_in[8];
    const float* Cubih = (const float*)d_in[9];
    const float* Cubhh = (const float*)d_in[10];
    const float* LuWih = (const float*)d_in[11];
    const float* LuWhh = (const float*)d_in[12];
    const float* Lubih = (const float*)d_in[13];
    const float* Lubhh = (const float*)d_in[14];
    const float* LvW1  = (const float*)d_in[15];
    const float* Lvb1  = (const float*)d_in[16];
    const float* LvW2  = (const float*)d_in[17];
    const float* Lvb2  = (const float*)d_in[18];
    // d_in[19] = T (device scalar, unreadable under graph capture) -> TSTEPS=16

    const long BL = (long)NBATCH * LDIM;  // 16384 literal rows
    const long BC = (long)NBATCH * CDIM;  // 32768 clause rows

    // Workspace carve-up (floats).
    float* ws  = (float*)d_ws;
    float* l   = ws;  ws += BL * HDIM;
    float* c   = ws;  ws += BC * HDIM;
    float* t0  = ws;  ws += BL * 256;          // covers BC*H too
    float* t1  = ws;  ws += BC * HDIM;
    float* mtl = ws;  ws += BC * HDIM;
    float* mc  = ws;  ws += BL * HDIM;
    float* gi  = ws;  ws += BC * 384;
    float* gh  = ws;  ws += BC * 384;
    float* gib = ws;  ws += 512;
    const size_t baseBytes = (size_t)(ws - (float*)d_ws) * sizeof(float);

    const long nMs = (long)NBATCH * LDIM * CDIM;            // 67,108,864
    const bool useHalfMs = ws_size >= baseBytes + (size_t)nMs * sizeof(_Float16);
    _Float16* Msh = (_Float16*)ws;

    // Init state + constants.
    { long t = BL * HDIM; fill_bcast<<<(unsigned)((t + 255) / 256), 256, 0, stream>>>(l, Linit, t); }
    { long t = BC * HDIM; fill_bcast<<<(unsigned)((t + 255) / 256), 256, 0, stream>>>(c, Cinit, t); }
    compute_gibase<<<2, 192, 0, stream>>>(Linit, LuWih, Lubih, gib);
    if (useHalfMs)
        f32_to_f16<<<(unsigned)(nMs / 4 / 256), 256, 0, stream>>>(Ms, Msh, nMs);

    const int H = HDIM;
    for (int t = 0; t < TSTEPS; ++t) {
        // Lmsg MLP (4 layers H->H, ReLU on first 3): l -> t1
        launch_gemm<float, false, false, true >(stream, l,  LmsgW + 0*H*H, Lmsgb + 0*H, t0, (int)BL, H, H, H, H, H);
        launch_gemm<float, false, false, true >(stream, t0, LmsgW + 1*H*H, Lmsgb + 1*H, t1, (int)BL, H, H, H, H, H);
        launch_gemm<float, false, false, true >(stream, t1, LmsgW + 2*H*H, Lmsgb + 2*H, t0, (int)BL, H, H, H, H, H);
        launch_gemm<float, false, false, false>(stream, t0, LmsgW + 3*H*H, Lmsgb + 3*H, t1, (int)BL, H, H, H, H, H);

        // MTL[b] = Ms[b]^T @ Lm[b] : M=C, N=H, K=L, A transposed, batched
        if (useHalfMs)
            launch_gemm<_Float16, true, false, false>(stream, Msh, t1, nullptr, mtl, CDIM, H, LDIM,
                CDIM, H, H, (long)LDIM*CDIM, (long)LDIM*H, (long)CDIM*H, NBATCH);
        else
            launch_gemm<float, true, false, false>(stream, Ms, t1, nullptr, mtl, CDIM, H, LDIM,
                CDIM, H, H, (long)LDIM*CDIM, (long)LDIM*H, (long)CDIM*H, NBATCH);

        // Clause GRU: gi = MTL @ Wih^T + bih ; gh = c @ Whh^T + bhh
        launch_gemm<float, false, true, false>(stream, mtl, CuWih, Cubih, gi, (int)BC, 384, H, H, H, 384);
        launch_gemm<float, false, true, false>(stream, c,   CuWhh, Cubhh, gh, (int)BC, 384, H, H, H, 384);
        { long n = BC * H; gru_combine<<<(unsigned)((n + 255) / 256), 256, 0, stream>>>(gi, gh, c, (int)BC); }

        // Cmsg MLP: c -> t1
        launch_gemm<float, false, false, true >(stream, c,  CmsgW + 0*H*H, Cmsgb + 0*H, t0, (int)BC, H, H, H, H, H);
        launch_gemm<float, false, false, true >(stream, t0, CmsgW + 1*H*H, Cmsgb + 1*H, t1, (int)BC, H, H, H, H, H);
        launch_gemm<float, false, false, true >(stream, t1, CmsgW + 2*H*H, Cmsgb + 2*H, t0, (int)BC, H, H, H, H, H);
        launch_gemm<float, false, false, false>(stream, t0, CmsgW + 3*H*H, Cmsgb + 3*H, t1, (int)BC, H, H, H, H, H);

        // MC[b] = Ms[b] @ Cm[b] : M=L, N=H, K=C, batched
        if (useHalfMs)
            launch_gemm<_Float16, false, false, false>(stream, Msh, t1, nullptr, mc, LDIM, H, CDIM,
                CDIM, H, H, (long)LDIM*CDIM, (long)CDIM*H, (long)LDIM*H, NBATCH);
        else
            launch_gemm<float, false, false, false>(stream, Ms, t1, nullptr, mc, LDIM, H, CDIM,
                CDIM, H, H, (long)LDIM*CDIM, (long)CDIM*H, (long)LDIM*H, NBATCH);

        // Literal GRU: gi = gi_base + MC @ Wih[:,H:]^T ; gh = l @ Whh^T + bhh
        launch_gemm<float, false, true, false>(stream, mc, LuWih + H, gib,   gi, (int)BL, 384, H, H, 256, 384);
        launch_gemm<float, false, true, false>(stream, l,  LuWhh,     Lubhh, gh, (int)BL, 384, H, H, H,   384);
        { long n = BL * H; gru_combine<<<(unsigned)((n + 255) / 256), 256, 0, stream>>>(gi, gh, l, (int)BL); }
    }

    // Readout: h1 = relu(l @ Lv_W1 + b1) ; lvotes = h1 @ Lv_W2 + b2 ; y = sigmoid(mean_L)
    launch_gemm<float, false, false, true>(stream, l, LvW1, Lvb1, t0, (int)BL, 256, H, H, 256, 256);
    float* y  = (float*)d_out;
    float* lv = y + NBATCH;
    lvotes_kernel<<<(unsigned)((BL + 255) / 256), 256, 0, stream>>>(t0, LvW2, Lvb2, lv, (int)BL);
    readout_mean<<<NBATCH, 256, 0, stream>>>(lv, y);
}
